// GraphMultisetTransformer_9294309228648
// MI455X (gfx1250) — matrix operand order repaired and checked by reading, not compiled, not run on hardware
//
#include <hip/hip_runtime.h>
#include <hip/hip_bf16.h>
#include <cmath>

typedef __attribute__((ext_vector_type(16))) _Float16 v16h;
typedef __attribute__((ext_vector_type(8)))  _Float16 v8h;
typedef __attribute__((ext_vector_type(8)))  float    v8f;

#define HID   128
#define HEADS 4
#define NB    32
#define NMAX  2048
#define SCALE 0.08838834764831845f   // 1/sqrt(128)
#define MSTRIP 4                     // M tiles per wave in projection GEMMs
#define KTILE  4                     // k tiles per wave in att_stats

// ---------------- WMMA fragment helpers (layouts per CDNA5 ISA 7.12.2) ----------------

// A-matrix 16x32 f16: lane holds row M=lane%16; element e -> K = (e/8)*16 + (lane/16)*8 + e%8
__device__ __forceinline__ v16h load_a_frag(const _Float16* base, int lda, int row, int k0, int lane) {
  const _Float16* p = base + (size_t)row * lda + k0 + ((lane >> 4) << 3);
  v8h lo = *(const v8h*)p;
  v8h hi = *(const v8h*)(p + 16);
  v16h a;
#pragma unroll
  for (int i = 0; i < 8; ++i) { a[i] = lo[i]; a[i + 8] = hi[i]; }
  return a;
}

// B-matrix 32x16 f16 from BT (=[N,K] row-major): lane holds col N=lane%16; element e -> K = (e/8)*16 + (lane/16)*8 + e%8
__device__ __forceinline__ v16h load_bt_frag(const _Float16* base, int ldbt, int nrow, int k0, int lane) {
  const _Float16* p = base + (size_t)nrow * ldbt + k0 + ((lane >> 4) << 3);
  v8h lo = *(const v8h*)p;
  v8h hi = *(const v8h*)(p + 16);
  v16h b;
#pragma unroll
  for (int i = 0; i < 8; ++i) { b[i] = lo[i]; b[i + 8] = hi[i]; }
  return b;
}

__device__ __forceinline__ v8f wmma32(v16h a, v16h b, v8f c) {
  return __builtin_amdgcn_wmma_f32_16x16x32_f16(false, a, false, b, (short)0, c, false, false);
}

// ---------------- small utility kernels ----------------

__global__ void k_starts(const int* __restrict__ batch, int T, int* __restrict__ starts) {
  int b = threadIdx.x;
  if (b >= NB) return;
  int lo = 0, hi = T;
  while (lo < hi) { int mid = (lo + hi) >> 1; if (batch[mid] < b) lo = mid + 1; else hi = mid; }
  starts[b] = lo;
}

__global__ void k_zero_f16(_Float16* __restrict__ p, long long n) {
  long long i = (long long)blockIdx.x * blockDim.x + threadIdx.x;
  long long st = (long long)gridDim.x * blockDim.x;
  for (; i < n; i += st) p[i] = (_Float16)0.0f;
}

__global__ void k_f32_to_f16(const float* __restrict__ s, _Float16* __restrict__ d, long long n) {
  long long i = (long long)blockIdx.x * blockDim.x + threadIdx.x;
  long long st = (long long)gridDim.x * blockDim.x;
  for (; i < n; i += st) d[i] = (_Float16)s[i];
}

__global__ void k_transpose_w(const float* __restrict__ src, _Float16* __restrict__ dst) {
  int i = blockIdx.x * blockDim.x + threadIdx.x;           // 128*128
  if (i >= HID * HID) return;
  int n = i >> 7, k = i & 127;
  dst[i] = (_Float16)src[k * HID + n];                     // dst[N][K] = w[K][N]
}

__global__ void k_s2pad(const float* __restrict__ s2, _Float16* __restrict__ dst) {
  int i = blockIdx.x * blockDim.x + threadIdx.x;           // 16*128
  if (i >= 16 * HID) return;
  dst[i] = (i >> 7) == 0 ? (_Float16)s2[i & 127] : (_Float16)0.0f;
}

__global__ void k_gather(const float* __restrict__ fin, float* __restrict__ out) {
  int i = blockIdx.x * blockDim.x + threadIdx.x;           // 32*128
  if (i >= NB * HID) return;
  out[i] = fin[(i >> 7) * (16 * HID) + (i & 127)];         // row 0 of each [16,128] block
}

// ---------------- generic projection GEMM: C[M,128] = A[M,128] @ W + bias ----------------
// One wave: fixed N-tile, strip of MSTRIP M-tiles; weight fragments held in registers.
// Output mode is compile-time: RESID -> C = A + relu(C); TOVT -> scatter to V^T [B,128,nk];
// F32OUT -> f32 row-major store; otherwise f16 row-major store.

template <bool RESID, bool TOVT, bool F32OUT>
__global__ void proj_gemm(const _Float16* __restrict__ A, int M,
                          const _Float16* __restrict__ WT, const float* __restrict__ bias,
                          _Float16* __restrict__ out16, float* __restrict__ out32,
                          _Float16* __restrict__ outVT, int nkVT) {
  int wave = (blockIdx.x * blockDim.x + threadIdx.x) >> 5;
  int lane = threadIdx.x & 31;
  int Mt = M >> 4;
  int nt = wave & 7, ms = wave >> 3;
  if (ms * MSTRIP >= Mt) return;
  int colB = nt << 4;
  int col  = colB + (lane & 15);
  float bvl = bias[col];

  v16h bfrag[4];
#pragma unroll
  for (int kk = 0; kk < 4; ++kk)
    bfrag[kk] = load_bt_frag(WT, HID, colB + (lane & 15), kk * 32, lane);

#pragma unroll
  for (int t = 0; t < MSTRIP; ++t) {
    int mt = ms * MSTRIP + t;
    if (mt >= Mt) break;
    int rowA = (mt << 4) + (lane & 15);
    v8f acc = {};
#pragma unroll
    for (int kk = 0; kk < 4; ++kk) {
      v16h a = load_a_frag(A, HID, rowA, kk * 32, lane);
      acc = wmma32(a, bfrag[kk], acc);
    }
#pragma unroll
    for (int i = 0; i < 8; ++i) {
      int row = (mt << 4) + i + ((lane >> 4) << 3);
      float v = acc[i] + bvl;
      if (RESID) {
        float r = (float)A[(size_t)row * HID + col];
        v = r + (v > 0.0f ? v : 0.0f);                      // out + relu(out@wo+bo)
      }
      if (TOVT) {
        int b_ = row / nkVT, rr = row % nkVT;
        outVT[((size_t)b_ * HID + col) * (size_t)nkVT + rr] = (_Float16)v; // V^T [B,128,nk]
      } else if (F32OUT) {
        out32[(size_t)row * HID + col] = v;
      } else {
        out16[(size_t)row * HID + col] = (_Float16)v;
      }
    }
  }
}

// lin1 + scatter-to-dense (to_dense_batch), same register blocking
__global__ void lin1_scatter(const _Float16* __restrict__ A, int M,
                             const _Float16* __restrict__ WT, const float* __restrict__ bias,
                             const int* __restrict__ batch, const int* __restrict__ starts,
                             _Float16* __restrict__ BX) {
  int wave = (blockIdx.x * blockDim.x + threadIdx.x) >> 5;
  int lane = threadIdx.x & 31;
  int Mt = M >> 4;
  int nt = wave & 7, ms = wave >> 3;
  if (ms * MSTRIP >= Mt) return;
  int colB = nt << 4;
  int col  = colB + (lane & 15);
  float bvl = bias[col];

  v16h bfrag[4];
#pragma unroll
  for (int kk = 0; kk < 4; ++kk)
    bfrag[kk] = load_bt_frag(WT, HID, colB + (lane & 15), kk * 32, lane);

#pragma unroll
  for (int t = 0; t < MSTRIP; ++t) {
    int mt = ms * MSTRIP + t;
    if (mt >= Mt) break;
    int rowA = (mt << 4) + (lane & 15);
    v8f acc = {};
#pragma unroll
    for (int kk = 0; kk < 4; ++kk) {
      v16h a = load_a_frag(A, HID, rowA, kk * 32, lane);
      acc = wmma32(a, bfrag[kk], acc);
    }
#pragma unroll
    for (int i = 0; i < 8; ++i) {
      int row = (mt << 4) + i + ((lane >> 4) << 3);
      int g = batch[row];
      int pos = row - starts[g];
      BX[((size_t)g * NMAX + pos) * HID + col] = (_Float16)(acc[i] + bvl);
    }
  }
}

// ---------------- attention pass 1: per-key-column softmax stats over the QUERY axis ----------------
// One wave owns KTILE (=4) 16-wide key strips; K fragments + online (m,d) held in registers.

__global__ void att_stats(const _Float16* __restrict__ Qp, long long qBstride,
                          const _Float16* __restrict__ Kp,
                          float* __restrict__ sm, float* __restrict__ sd,
                          int nq, int nq_valid, int nk) {
  int wave = (blockIdx.x * blockDim.x + threadIdx.x) >> 5;
  int lane = threadIdx.x & 31;
  int kgroups = nk >> 6;                                    // 64 keys per wave
  int kg = wave % kgroups, bh = wave / kgroups;
  if (bh >= NB * HEADS) return;
  int h = bh & 3, b = bh >> 2;
  const _Float16* Kb = Kp + (size_t)b * nk * HID;
  const _Float16* Qb = Qp + (size_t)b * qBstride;

  v16h bk[KTILE];
#pragma unroll
  for (int j = 0; j < KTILE; ++j)
    bk[j] = load_bt_frag(Kb, HID, (kg * 64) + j * 16 + (lane & 15), h * 32, lane);

  float m[KTILE], d[KTILE];
#pragma unroll
  for (int j = 0; j < KTILE; ++j) { m[j] = -__builtin_inff(); d[j] = 0.0f; }

  for (int qt = 0; qt < nq; qt += 16) {
    v16h aq = load_a_frag(Qb, HID, qt + (lane & 15), h * 32, lane);
#pragma unroll
    for (int j = 0; j < KTILE; ++j) {
      v8f c = {};
      c = wmma32(aq, bk[j], c);                             // scores tile [16q x 16k]
#pragma unroll
      for (int i = 0; i < 8; ++i) {
        int row = qt + i + ((lane >> 4) << 3);
        if (row < nq_valid) {
          float s = c[i] * SCALE;
          float nm = fmaxf(m[j], s);
          d[j] = d[j] * __expf(m[j] - nm) + __expf(s - nm);
          m[j] = nm;
        }
      }
    }
  }
#pragma unroll
  for (int j = 0; j < KTILE; ++j) {
    float om = __shfl_xor(m[j], 16, 32);
    float od = __shfl_xor(d[j], 16, 32);
    float nm = fmaxf(m[j], om);
    float dd = d[j] * __expf(m[j] - nm) + od * __expf(om - nm);
    if (lane < 16) {
      size_t idx = (size_t)bh * nk + kg * 64 + j * 16 + lane;
      sm[idx] = nm;
      sd[idx] = dd;
    }
  }
}

// ---------------- attention pass 2: out = Qh + softmax_q(S) @ Vh ----------------
// One wave owns one q-tile and BOTH 16-wide d-strips of its head: scores computed once per k-step.

__global__ void att_out(const _Float16* __restrict__ Qp, long long qBstride,
                        const _Float16* __restrict__ Kp, const _Float16* __restrict__ VT,
                        const float* __restrict__ sm, const float* __restrict__ sd,
                        _Float16* __restrict__ out, int nq, int nk) {
  __shared__ __align__(32) _Float16 stage[8][16 * 32];
  int wave = (blockIdx.x * blockDim.x + threadIdx.x) >> 5;
  int lane = threadIdx.x & 31;
  int wv = threadIdx.x >> 5;
  int qtiles = nq >> 4;
  if (wave >= NB * HEADS * qtiles) return;
  int qt = (wave % qtiles) << 4;
  int bh = wave / qtiles;
  int h = bh & 3, b = bh >> 2;
  const _Float16* Qb  = Qp + (size_t)b * qBstride;
  const _Float16* Kb  = Kp + (size_t)b * nk * HID;
  const _Float16* Vb0 = VT + ((size_t)b * HID + h * 32) * (size_t)nk;       // V^T rows d=0..15
  const _Float16* Vb1 = Vb0 + (size_t)16 * nk;                              // V^T rows d=16..31
  const float* smb = sm + (size_t)bh * nk;
  const float* sdb = sd + (size_t)bh * nk;
  _Float16* st = &stage[wv][0];

  v16h aq = load_a_frag(Qb, HID, qt + (lane & 15), h * 32, lane);           // Qh tile (constant)
  v8f acc0 = {}, acc1 = {};
  for (int k0 = 0; k0 < nk; k0 += 32) {
#pragma unroll
    for (int tt = 0; tt < 2; ++tt) {
      int kt0 = k0 + tt * 16;
      v16h bkf = load_bt_frag(Kb, HID, kt0 + (lane & 15), h * 32, lane);
      v8f c = {};
      c = wmma32(aq, bkf, c);                                // recompute scores tile
      int kcol = kt0 + (lane & 15);
      float mm = smb[kcol];
      float inv = 1.0f / sdb[kcol];
#pragma unroll
      for (int i = 0; i < 8; ++i) {
        int ql = i + ((lane >> 4) << 3);
        float w = __expf(c[i] * SCALE - mm) * inv;           // normalized over query axis
        st[ql * 32 + tt * 16 + (lane & 15)] = (_Float16)w;   // C-layout -> row-major 16x32 tile
      }
    }
    asm volatile("" ::: "memory");                           // keep LDS store/load order (same-wave, in-order DS)
    v16h as = load_a_frag(st, 32, (lane & 15), 0, lane);     // reload normalized tile as A-operand
    asm volatile("" ::: "memory");
    v16h bv0 = load_bt_frag(Vb0, nk, (lane & 15), k0, lane); // Vh tiles from V^T (contiguous)
    acc0 = wmma32(as, bv0, acc0);
    v16h bv1 = load_bt_frag(Vb1, nk, (lane & 15), k0, lane);
    acc1 = wmma32(as, bv1, acc1);
  }
  int col0 = h * 32 + (lane & 15);
#pragma unroll
  for (int i = 0; i < 8; ++i) {
    int row = qt + i + ((lane >> 4) << 3);
    float v0 = acc0[i] + (float)Qb[(size_t)row * HID + col0];        // + Qh residual
    float v1 = acc1[i] + (float)Qb[(size_t)row * HID + col0 + 16];
    out[((size_t)b * nq + row) * HID + col0]      = (_Float16)v0;
    out[((size_t)b * nq + row) * HID + col0 + 16] = (_Float16)v1;
  }
}

// ---------------- host orchestration ----------------

static inline int blocks_for_waves(long long waves) { return (int)((waves * 32 + 255) / 256); }
static inline long long proj_waves(int M) { long long Mt = M >> 4; return ((Mt + MSTRIP - 1) / MSTRIP) * 8; }

extern "C" void kernel_launch(void* const* d_in, const int* in_sizes, int n_in,
                              void* d_out, int out_size, void* d_ws, size_t ws_size,
                              hipStream_t stream) {
  const int T   = in_sizes[0] / HID;       // 32768
  const int S0N = in_sizes[4] / HID;       // 512 seeds

  const float* x      = (const float*)d_in[0];
  const int*   batch  = (const int*)d_in[1];
  const float* S0     = (const float*)d_in[4];
  const float* S2     = (const float*)d_in[5];

  // weight/bias input indices: lin1=(2,3); per MAB m: wq,bq,wk,bk,wv,bv,wo,bo
  const int wIdx[13] = {2, 6, 8, 10, 12, 14, 16, 18, 20, 22, 24, 26, 28};
  const float* bias_lin1 = (const float*)d_in[3];
  const float* bq[3] = {(const float*)d_in[7],  (const float*)d_in[15], (const float*)d_in[23]};
  const float* bk[3] = {(const float*)d_in[9],  (const float*)d_in[17], (const float*)d_in[25]};
  const float* bv[3] = {(const float*)d_in[11], (const float*)d_in[19], (const float*)d_in[27]};
  const float* bo[3] = {(const float*)d_in[13], (const float*)d_in[21], (const float*)d_in[29]};

  // ---- workspace layout (bump allocator, 256B aligned) ----
  char* ws = (char*)d_ws;
  size_t off = 0;
  auto alloc = [&](size_t bytes) { off = (off + 255) & ~(size_t)255; size_t o = off; off += bytes; return o; };
  _Float16* XF16 = (_Float16*)(ws + alloc((size_t)T * HID * 2));
  _Float16* WT   = (_Float16*)(ws + alloc((size_t)13 * HID * HID * 2));
  _Float16* S0F  = (_Float16*)(ws + alloc((size_t)S0N * HID * 2));
  _Float16* S2P  = (_Float16*)(ws + alloc((size_t)16 * HID * 2));
  _Float16* QP0  = (_Float16*)(ws + alloc((size_t)S0N * HID * 2));
  _Float16* QP1  = (_Float16*)(ws + alloc((size_t)NB * S0N * HID * 2));
  _Float16* QP2  = (_Float16*)(ws + alloc((size_t)16 * HID * 2));
  _Float16* BX   = (_Float16*)(ws + alloc((size_t)NB * NMAX * HID * 2));
  _Float16* KP   = (_Float16*)(ws + alloc((size_t)NB * NMAX * HID * 2));
  _Float16* VT   = (_Float16*)(ws + alloc((size_t)NB * HID * NMAX * 2));
  float*    SM   = (float*)   (ws + alloc((size_t)NB * HEADS * NMAX * 4));
  float*    SD   = (float*)   (ws + alloc((size_t)NB * HEADS * NMAX * 4));
  _Float16* AO   = (_Float16*)(ws + alloc((size_t)NB * S0N * HID * 2));
  _Float16* X1   = (_Float16*)(ws + alloc((size_t)NB * S0N * HID * 2));
  _Float16* X2   = (_Float16*)(ws + alloc((size_t)NB * S0N * HID * 2));
  float*    FIN  = (float*)   (ws + alloc((size_t)NB * 16 * HID * 4));
  int*      ST   = (int*)     (ws + alloc((size_t)64 * 4));
  (void)ws_size; (void)n_in; (void)out_size;

  // ---- prep ----
  k_starts<<<1, 32, 0, stream>>>(batch, T, ST);
  k_zero_f16<<<1024, 256, 0, stream>>>(BX, (long long)NB * NMAX * HID);
  for (int i = 0; i < 13; ++i)
    k_transpose_w<<<64, 256, 0, stream>>>((const float*)d_in[wIdx[i]], WT + (size_t)i * HID * HID);
  k_f32_to_f16<<<1024, 256, 0, stream>>>(x, XF16, (long long)T * HID);
  k_f32_to_f16<<<256, 256, 0, stream>>>(S0, S0F, (long long)S0N * HID);
  k_s2pad<<<8, 256, 0, stream>>>(S2, S2P);

  // ---- lin1 + to_dense_batch ----
  lin1_scatter<<<blocks_for_waves(proj_waves(T)), 256, 0, stream>>>(
      XF16, T, WT + 0 * HID * HID, bias_lin1, batch, ST, BX);

  // ---- MAB0: PMA(S0 seeds, dense bx) ----
  proj_gemm<false, false, false><<<blocks_for_waves(proj_waves(S0N)), 256, 0, stream>>>(
      S0F, S0N, WT + 1 * HID * HID, bq[0], QP0, nullptr, nullptr, 1);                       // Qp (shared over b)
  proj_gemm<false, false, false><<<blocks_for_waves(proj_waves(NB * NMAX)), 256, 0, stream>>>(
      BX, NB * NMAX, WT + 2 * HID * HID, bk[0], KP, nullptr, nullptr, 1);                   // Kp
  proj_gemm<false, true, false><<<blocks_for_waves(proj_waves(NB * NMAX)), 256, 0, stream>>>(
      BX, NB * NMAX, WT + 3 * HID * HID, bv[0], nullptr, nullptr, VT, NMAX);                // V^T
  att_stats<<<blocks_for_waves((long long)NB * HEADS * (NMAX / 64)), 256, 0, stream>>>(
      QP0, 0LL, KP, SM, SD, S0N, S0N, NMAX);
  att_out<<<blocks_for_waves((long long)NB * HEADS * (S0N / 16)), 256, 0, stream>>>(
      QP0, 0LL, KP, VT, SM, SD, AO, S0N, NMAX);
  proj_gemm<true, false, false><<<blocks_for_waves(proj_waves(NB * S0N)), 256, 0, stream>>>(
      AO, NB * S0N, WT + 4 * HID * HID, bo[0], X1, nullptr, nullptr, 1);                    // + relu(.@wo+bo)

  // ---- MAB1: SAB(x1, x1) ----
  proj_gemm<false, false, false><<<blocks_for_waves(proj_waves(NB * S0N)), 256, 0, stream>>>(
      X1, NB * S0N, WT + 5 * HID * HID, bq[1], QP1, nullptr, nullptr, 1);
  proj_gemm<false, false, false><<<blocks_for_waves(proj_waves(NB * S0N)), 256, 0, stream>>>(
      X1, NB * S0N, WT + 6 * HID * HID, bk[1], KP, nullptr, nullptr, 1);
  proj_gemm<false, true, false><<<blocks_for_waves(proj_waves(NB * S0N)), 256, 0, stream>>>(
      X1, NB * S0N, WT + 7 * HID * HID, bv[1], nullptr, nullptr, VT, S0N);
  att_stats<<<blocks_for_waves((long long)NB * HEADS * (S0N / 64)), 256, 0, stream>>>(
      QP1, (long long)S0N * HID, KP, SM, SD, S0N, S0N, S0N);
  att_out<<<blocks_for_waves((long long)NB * HEADS * (S0N / 16)), 256, 0, stream>>>(
      QP1, (long long)S0N * HID, KP, VT, SM, SD, AO, S0N, S0N);
  proj_gemm<true, false, false><<<blocks_for_waves(proj_waves(NB * S0N)), 256, 0, stream>>>(
      AO, NB * S0N, WT + 8 * HID * HID, bo[1], X2, nullptr, nullptr, 1);

  // ---- MAB2: PMA(1 seed, x2); q padded to 16, softmax masked to nq_valid=1 ----
  proj_gemm<false, false, false><<<1, 256, 0, stream>>>(
      S2P, 16, WT + 9 * HID * HID, bq[2], QP2, nullptr, nullptr, 1);
  proj_gemm<false, false, false><<<blocks_for_waves(proj_waves(NB * S0N)), 256, 0, stream>>>(
      X2, NB * S0N, WT + 10 * HID * HID, bk[2], KP, nullptr, nullptr, 1);
  proj_gemm<false, true, false><<<blocks_for_waves(proj_waves(NB * S0N)), 256, 0, stream>>>(
      X2, NB * S0N, WT + 11 * HID * HID, bv[2], nullptr, nullptr, VT, S0N);
  att_stats<<<blocks_for_waves((long long)NB * HEADS * (S0N / 64)), 256, 0, stream>>>(
      QP2, 0LL, KP, SM, SD, 16, 1, S0N);
  att_out<<<blocks_for_waves((long long)NB * HEADS * 1), 256, 0, stream>>>(
      QP2, 0LL, KP, VT, SM, SD, AO, 16, S0N);
  proj_gemm<true, false, true><<<blocks_for_waves(proj_waves(NB * 16)), 256, 0, stream>>>(
      AO, NB * 16, WT + 12 * HID * HID, bo[2], nullptr, FIN, nullptr, 1);

  // ---- squeeze(1): take row 0 of each graph ----
  k_gather<<<16, 256, 0, stream>>>(FIN, (float*)d_out);
}
